// Model_49933289783344
// MI455X (gfx1250) — matrix operand order repaired and checked by reading, not compiled, not run on hardware
//
#include <hip/hip_runtime.h>

// Problem sizes
#define B_  256
#define S_  336
#define C_  321
#define H_  512
#define P_  96
#define CP  336   // C padded to 21 tiles of 16
#define SP  352   // S padded to 11 k-steps of 32

typedef __attribute__((ext_vector_type(16))) _Float16    v16h;
typedef __attribute__((ext_vector_type(8)))  float        v8f;
typedef __attribute__((ext_vector_type(4)))  unsigned int v4u;

union Frag { v16h h; v4u u[2]; };

// Build a 16-element f16 fragment from two 16-byte chunks.
__device__ __forceinline__ v16h load_frag(const _Float16* p0, const _Float16* p1) {
    Frag f;
    f.u[0] = *(const v4u*)p0;
    f.u[1] = *(const v4u*)p1;
    return f.h;
}

__device__ __forceinline__ v8f wmma_f16(v16h a, v16h b, v8f c) {
    return __builtin_amdgcn_wmma_f32_16x16x32_f16(false, a, false, b, (short)0, c,
                                                  false, false);
}

// ---------------- init: f32 -> f16 weight conversion (with column padding) ----
__global__ void k_convert_pad(_Float16* __restrict__ dst, const float* __restrict__ src,
                              int rows, int scols, int dcols) {
    int n = rows * dcols;
    for (int idx = blockIdx.x * blockDim.x + threadIdx.x; idx < n;
         idx += gridDim.x * blockDim.x) {
        int r = idx / dcols, c = idx - r * dcols;
        dst[idx] = (c < scols) ? (_Float16)src[r * scols + c] : (_Float16)0.f;
    }
}

__global__ void k_zero_f16(_Float16* __restrict__ dst, int n) {
    for (int idx = blockIdx.x * blockDim.x + threadIdx.x; idx < n;
         idx += gridDim.x * blockDim.x)
        dst[idx] = (_Float16)0.f;
}

// ---------------- kernel 1: IT[b,c,h] = x[b,:,c] @ Wi.T + bi --------------------
// Block: 128 threads (4 waves). Block computes one 16-row c-tile x full H=512.
// Each wave owns 8 N-tiles. A tile (16c x 32s, strided in global) is staged in
// LDS in WMMA-fragment order; B fragments come straight from the f16 Wi rows.
__global__ __launch_bounds__(128) void k_input_gemm(
    const float* __restrict__ x, const float* __restrict__ bi,
    const _Float16* __restrict__ Wif, _Float16* __restrict__ IT) {
    __shared__ _Float16 lds_a[32 * 16];   // [lane][j] fragment layout, 1 KB

    const int ct   = blockIdx.x;          // 0..20  (c tile)
    const int b    = blockIdx.y;          // 0..255 (batch = scan step)
    const int tid  = threadIdx.x;
    const int lane = tid & 31;
    const int wave = tid >> 5;            // 0..3
    const int c0   = ct * 16;
    const int lo   = lane & 15, mid = lane >> 4;

    v8f acc[8];
    const v8f vzero = {0.f, 0.f, 0.f, 0.f, 0.f, 0.f, 0.f, 0.f};
    for (int i = 0; i < 8; ++i) acc[i] = vzero;

    for (int kk = 0; kk < 11; ++kk) {
        const int s0 = kk * 32;
        // Cooperative A-tile load: x[b, s0+k, c0+c] (coalesced along c), placed
        // directly in A-fragment order: lane = c + 16*((k>>3)&1), j = 8*(k>>4) + (k&7)
        for (int i = 0; i < 4; ++i) {
            int idx = tid + i * 128;          // 0..511
            int k = idx >> 4;                 // s offset in tile, 0..31
            int c = idx & 15;
            int s = s0 + k, cc = c0 + c;
            float v = (s < S_ && cc < C_) ? x[((size_t)b * S_ + s) * C_ + cc] : 0.f;
            int hi = k >> 4, md = (k >> 3) & 1, lw = k & 7;
            lds_a[(c + 16 * md) * 16 + (8 * hi + lw)] = (_Float16)v;
        }
        __syncthreads();
        v16h afrag = load_frag(&lds_a[lane * 16], &lds_a[lane * 16 + 8]);
        for (int nt = 0; nt < 8; ++nt) {
            int hb = (wave * 8 + nt) * 16;
            const _Float16* bp = Wif + (size_t)(hb + lo) * SP + s0 + 8 * mid;
            v16h bfrag = load_frag(bp, bp + 16);
            acc[nt] = wmma_f16(afrag, bfrag, acc[nt]);
        }
        __syncthreads();
    }
    // epilogue: + bi, store f16 (pad rows c>=321 get harmless values; rows independent)
    for (int nt = 0; nt < 8; ++nt) {
        int hidx = (wave * 8 + nt) * 16 + lo;
        float bv = bi[hidx];
        for (int r = 0; r < 8; ++r) {
            int c = c0 + r + 8 * mid;
            IT[((size_t)b * CP + c) * H_ + hidx] = (_Float16)(acc[nt][r] + bv);
        }
    }
}

// ---------------- kernel 2: one recurrence step ---------------------------------
// hh = h_prev @ Wh.T + bh ; a = relu(it + hh) ; h_new = relu(a + hh)
// h_new overwrites IT[t] in place. One wave per 16x16 tile; grid (21, 32).
__global__ __launch_bounds__(32) void k_recurrence(
    const _Float16* __restrict__ hprev, const float* __restrict__ bh,
    const _Float16* __restrict__ Whf, _Float16* __restrict__ it_t) {
    const int m = blockIdx.x;             // c tile 0..20
    const int n = blockIdx.y;             // h tile 0..31
    const int lane = threadIdx.x & 31;
    const int lo = lane & 15, mid = lane >> 4;
    const int c0 = m * 16, h0 = n * 16;

    v8f acc = {0.f, 0.f, 0.f, 0.f, 0.f, 0.f, 0.f, 0.f};
#pragma unroll 4
    for (int kk = 0; kk < 16; ++kk) {
        const int k0 = kk * 32;
        const _Float16* ap = hprev + (size_t)(c0 + lo) * H_ + k0 + 8 * mid;
        v16h afrag = load_frag(ap, ap + 16);
        const _Float16* bp = Whf + (size_t)(h0 + lo) * H_ + k0 + 8 * mid;
        v16h bfrag = load_frag(bp, bp + 16);
        acc = wmma_f16(afrag, bfrag, acc);
    }
    const int hidx = h0 + lo;
    const float bhv = bh[hidx];
    for (int r = 0; r < 8; ++r) {
        int c = c0 + r + 8 * mid;
        size_t off = (size_t)c * H_ + hidx;
        float hh = acc[r] + bhv;
        float a  = (float)it_t[off] + hh;
        a = a > 0.f ? a : 0.f;
        float hn = a + hh;
        hn = hn > 0.f ? hn : 0.f;
        it_t[off] = (_Float16)hn;
    }
}

// ---------------- kernel 3: out = (H_all @ Wf.T + bf >= 0) ? 1 : 0 --------------
// Rows = (b, c) flattened over the state buffer; one wave per 16-row tile x all
// 6 P-tiles. Output layout [B, P, C] float.
__global__ __launch_bounds__(32) void k_final(
    const _Float16* __restrict__ OUTS, const float* __restrict__ bf,
    const _Float16* __restrict__ Wff, float* __restrict__ out) {
    const int mt = blockIdx.x;            // 0..5375
    const int b = mt / 21, ct = mt - b * 21;
    const int c0 = ct * 16;
    const int lane = threadIdx.x & 31;
    const int lo = lane & 15, mid = lane >> 4;
    const size_t rowbase = (size_t)b * CP + c0;

    v8f acc[6];
    const v8f vzero = {0.f, 0.f, 0.f, 0.f, 0.f, 0.f, 0.f, 0.f};
    for (int i = 0; i < 6; ++i) acc[i] = vzero;

#pragma unroll 2
    for (int kk = 0; kk < 16; ++kk) {
        const int k0 = kk * 32;
        const _Float16* ap = OUTS + (rowbase + lo) * H_ + k0 + 8 * mid;
        v16h afrag = load_frag(ap, ap + 16);
        for (int nt = 0; nt < 6; ++nt) {
            const _Float16* bp = Wff + (size_t)(nt * 16 + lo) * H_ + k0 + 8 * mid;
            v16h bfrag = load_frag(bp, bp + 16);
            acc[nt] = wmma_f16(afrag, bfrag, acc[nt]);
        }
    }
    for (int nt = 0; nt < 6; ++nt) {
        int p = nt * 16 + lo;
        float bfv = bf[p];
        for (int r = 0; r < 8; ++r) {
            int c = c0 + r + 8 * mid;
            if (c < C_)
                out[((size_t)b * P_ + p) * C_ + c] =
                    (acc[nt][r] + bfv) >= 0.f ? 1.0f : 0.0f;
        }
    }
}

// ---------------- host side -----------------------------------------------------
extern "C" void kernel_launch(void* const* d_in, const int* in_sizes, int n_in,
                              void* d_out, int out_size, void* d_ws, size_t ws_size,
                              hipStream_t stream) {
    const float* x  = (const float*)d_in[0];
    const float* Wi = (const float*)d_in[1];
    const float* bi = (const float*)d_in[2];
    const float* Wh = (const float*)d_in[3];
    const float* bh = (const float*)d_in[4];
    const float* Wf = (const float*)d_in[5];
    const float* bf = (const float*)d_in[6];
    float* out = (float*)d_out;

    // workspace layout (f16):
    //   IT/state buffer [B][CP][H], Wh f16, Wi f16 (padded to SP), Wf f16, h0 zeros
    char* base = (char*)d_ws;
    constexpr size_t IT_BYTES = (size_t)B_ * CP * H_ * 2;        // 88,080,384
    constexpr size_t WH_BYTES = (size_t)H_ * H_ * 2;             //    524,288
    constexpr size_t WI_BYTES = (size_t)H_ * SP * 2;             //    360,448
    constexpr size_t WF_BYTES = (size_t)P_ * H_ * 2;             //     98,304
    _Float16* IT  = (_Float16*)(base);
    _Float16* WHf = (_Float16*)(base + IT_BYTES);
    _Float16* WIf = (_Float16*)(base + IT_BYTES + WH_BYTES);
    _Float16* WFf = (_Float16*)(base + IT_BYTES + WH_BYTES + WI_BYTES);
    _Float16* H0  = (_Float16*)(base + IT_BYTES + WH_BYTES + WI_BYTES + WF_BYTES);

    k_convert_pad<<<512, 256, 0, stream>>>(WHf, Wh, H_, H_, H_);
    k_convert_pad<<<512, 256, 0, stream>>>(WIf, Wi, H_, S_, SP);
    k_convert_pad<<<96, 256, 0, stream>>>(WFf, Wf, P_, H_, H_);
    k_zero_f16<<<336, 256, 0, stream>>>(H0, CP * H_);

    // Parallel input projection for all scan steps at once.
    k_input_gemm<<<dim3(21, B_), 128, 0, stream>>>(x, bi, WIf, IT);

    // Sequential recurrence over the batch dim; h_t overwrites IT[t] in place.
    for (int t = 0; t < B_; ++t) {
        const _Float16* hp = (t == 0) ? H0 : IT + (size_t)(t - 1) * CP * H_;
        k_recurrence<<<dim3(21, 32), 32, 0, stream>>>(hp, bh, WHf,
                                                      IT + (size_t)t * CP * H_);
    }

    // Fused projection + sign threshold (sigmoid>=0.5 <=> logit>=0).
    k_final<<<dim3(B_ * 21), 32, 0, stream>>>(IT, bf, WFf, out);
}
